// TemporalSSMBlock_7009386627572
// MI455X (gfx1250) — matrix operand order repaired and checked by reading, not compiled, not run on hardware
//
#include <hip/hip_runtime.h>
#include <hip/hip_bf16.h>
#include <math.h>

#define BATCH 2
#define SEQ   1024
#define DM    1024
#define DI    2048
#define NST   16
#define DTR   64
#define XDW   96            // dt_rank + 2*N = 64 + 32
#define MROWS (BATCH*SEQ)   // 2048

typedef __bf16 bf16x16 __attribute__((ext_vector_type(16)));
typedef float  f32x8   __attribute__((ext_vector_type(8)));

union Frag16 { bf16x16 v; uint4 q[2]; };

__device__ __forceinline__ unsigned short f2bf(float f) {
  unsigned int u = __float_as_uint(f);
  unsigned int r = 0x7FFFu + ((u >> 16) & 1u);
  return (unsigned short)((u + r) >> 16);
}
__device__ __forceinline__ float silu(float x) { return x / (1.0f + __expf(-x)); }

// ---------------------------------------------------------------------------
// bf16 WMMA GEMM, register-tiled: each wave computes a (TM*16) x (TN*16)
// block of C[M,N] = A[M,K] (bf16 row-major) * Bt[N,K]^T (bf16).
// K % 32 == 0, M % (16*TM) == 0, N % (16*TN) == 0.
// Fragment layouts per CDNA5 ISA 7.12.2 (wave32):
//   A 16x32: lane l -> row l&15, K-chunks [k+8*(l>>4),+8) and [k+16+8*(l>>4),+8)
//   B 32x16 (from Bt rows): lane l -> col l&15, K-chunks [k+8*(l>>4),+8) and [k+16+8*(l>>4),+8)
//   D: vgpr r -> C[m0 + 8*(l>>4) + r, n0 + (l&15)]
// ---------------------------------------------------------------------------
template <int TM, int TN>
__global__ __launch_bounds__(256)
void wmma_gemm_tiled(const unsigned short* __restrict__ A,
                     const unsigned short* __restrict__ Bt,
                     float* __restrict__ C, int M, int N, int K) {
  const int lane = threadIdx.x & 31;
  const int wave = threadIdx.x >> 5;
  const int bn = N / (16 * TN);
  const int bm = M / (16 * TM);
  const int tile = blockIdx.x * 8 + wave;
  if (tile >= bm * bn) return;
  const int tm = tile / bn;
  const int tn = tile - tm * bn;

  const int hi = lane >> 4;
  const int mr = lane & 15;

  const unsigned short* Ab[TM];
  const unsigned short* Bb[TN];
#pragma unroll
  for (int r = 0; r < TM; ++r)
    Ab[r] = A + (size_t)(tm * TM * 16 + r * 16 + mr) * K + hi * 8;
#pragma unroll
  for (int c = 0; c < TN; ++c)
    Bb[c] = Bt + (size_t)(tn * TN * 16 + c * 16 + mr) * K + hi * 8;

  f32x8 acc[TM][TN] = {};
  for (int k = 0; k < K; k += 32) {
    Frag16 a[TM], b[TN];
#pragma unroll
    for (int r = 0; r < TM; ++r) {
      a[r].q[0] = *reinterpret_cast<const uint4*>(Ab[r] + k);
      a[r].q[1] = *reinterpret_cast<const uint4*>(Ab[r] + k + 16);
    }
#pragma unroll
    for (int c = 0; c < TN; ++c) {
      b[c].q[0] = *reinterpret_cast<const uint4*>(Bb[c] + k);
      b[c].q[1] = *reinterpret_cast<const uint4*>(Bb[c] + k + 16);
    }
#pragma unroll
    for (int r = 0; r < TM; ++r)
#pragma unroll
      for (int c = 0; c < TN; ++c)
        acc[r][c] = __builtin_amdgcn_wmma_f32_16x16x32_bf16(
            false, a[r].v, false, b[c].v, (short)0, acc[r][c], false, false);
  }

#pragma unroll
  for (int r = 0; r < TM; ++r) {
#pragma unroll
    for (int c = 0; c < TN; ++c) {
      float* Crow = C + (size_t)(tm * TM * 16 + r * 16 + hi * 8) * N +
                    tn * TN * 16 + c * 16 + mr;
#pragma unroll
      for (int v = 0; v < 8; ++v) Crow[(size_t)v * N] = acc[r][c][v];
    }
  }
}

// ---------------------------------------------------------------------------
// Weight convert + transpose: in f32 [K,N]  ->  out bf16 [N,K]
// ---------------------------------------------------------------------------
__global__ __launch_bounds__(256)
void transpose_f32_bf16(const float* __restrict__ in, unsigned short* __restrict__ out,
                        int K, int N) {
  size_t i = (size_t)blockIdx.x * 256 + threadIdx.x;
  if (i >= (size_t)K * N) return;
  int k = (int)(i / N), n = (int)(i - (size_t)k * N);
  out[(size_t)n * K + k] = f2bf(in[i]);
}

// ---------------------------------------------------------------------------
// LayerNorm over D_MODEL, output bf16.  One block per row.
// ---------------------------------------------------------------------------
__global__ __launch_bounds__(256)
void layernorm_bf16(const float* __restrict__ x, const float* __restrict__ gamma,
                    const float* __restrict__ beta, unsigned short* __restrict__ out) {
  const int row = blockIdx.x, tid = threadIdx.x;
  const float* xr = x + (size_t)row * DM;
  float v[4], s = 0.f, s2 = 0.f;
#pragma unroll
  for (int j = 0; j < 4; ++j) { v[j] = xr[tid + j * 256]; s += v[j]; s2 += v[j] * v[j]; }
  __shared__ float rs[256], rq[256];
  rs[tid] = s; rq[tid] = s2; __syncthreads();
  for (int off = 128; off > 0; off >>= 1) {
    if (tid < off) { rs[tid] += rs[tid + off]; rq[tid] += rq[tid + off]; }
    __syncthreads();
  }
  const float mu = rs[0] * (1.0f / DM);
  const float var = rq[0] * (1.0f / DM) - mu * mu;
  const float rstd = rsqrtf(var + 1e-5f);
  unsigned short* orow = out + (size_t)row * DM;
#pragma unroll
  for (int j = 0; j < 4; ++j) {
    int c = tid + j * 256;
    orow[c] = f2bf((v[j] - mu) * rstd * gamma[c] + beta[c]);
  }
}

// ---------------------------------------------------------------------------
// Causal depthwise conv (K=4) + SiLU.  dir=1 operates on time-reversed xs.
// xs lives in xz[:, 0:DI].  Outputs scan-time-ordered f32 + bf16 copies.
// ---------------------------------------------------------------------------
__global__ __launch_bounds__(256)
void conv_silu(const float* __restrict__ xz, const float* __restrict__ cw,
               const float* __restrict__ cb, float* __restrict__ xc_f,
               unsigned short* __restrict__ xc_b, int dir) {
  size_t i = (size_t)blockIdx.x * 256 + threadIdx.x;   // over MROWS*DI
  const int d = (int)(i % DI);
  const size_t row = i / DI;
  const int t = (int)(row % SEQ);
  const int b = (int)(row / SEQ);
  float acc = cb[d];
#pragma unroll
  for (int j = 0; j < 4; ++j) {
    int tt = t - 3 + j;
    if (tt >= 0) {
      int tr = dir ? (SEQ - 1 - tt) : tt;
      acc += cw[d * 4 + j] * xz[((size_t)b * SEQ + tr) * (2 * DI) + d];
    }
  }
  acc = silu(acc);
  xc_f[i] = acc;
  xc_b[i] = f2bf(acc);
}

// dt slice of x_dbl -> bf16 [M, 64]
__global__ __launch_bounds__(256)
void slice_dt_bf16(const float* __restrict__ xdbl, unsigned short* __restrict__ dtA) {
  size_t i = (size_t)blockIdx.x * 256 + threadIdx.x;   // over MROWS*DTR
  size_t m = i / DTR; int j = (int)(i - m * DTR);
  dtA[i] = f2bf(xdbl[m * XDW + j]);
}

// dt = softplus(dt_lin + dt_b), in place on [M, DI]
__global__ __launch_bounds__(256)
void bias_softplus(float* __restrict__ dtf, const float* __restrict__ dtb) {
  size_t i = (size_t)blockIdx.x * 256 + threadIdx.x;
  int d = (int)(i % DI);
  float v = dtf[i] + dtb[d];
  dtf[i] = (v > 20.f) ? v : log1pf(__expf(v));
}

// ---------------------------------------------------------------------------
// Selective-scan.  grid = (DI/64, BATCH); 64 threads; one d per thread.
// Writes y (+ skip) as bf16 into y_cat[:, col_off + d] at real-time index.
// ---------------------------------------------------------------------------
__global__ __launch_bounds__(64)
void ssm_scan(const float* __restrict__ xdbl, const float* __restrict__ dtf,
              const float* __restrict__ xc, const float* __restrict__ A_log,
              const float* __restrict__ D_skip, unsigned short* __restrict__ y_cat,
              int dir, int col_off) {
  const int b = blockIdx.y;
  const int d = blockIdx.x * 64 + threadIdx.x;
  float Arow[NST], h[NST];
#pragma unroll
  for (int n = 0; n < NST; ++n) { Arow[n] = -__expf(A_log[d * NST + n]); h[n] = 0.f; }
  const float Dv = D_skip[d];
  __shared__ float BC[2 * NST];
  for (int t = 0; t < SEQ; ++t) {
    const size_t row = (size_t)b * SEQ + t;
    if (threadIdx.x < 2 * NST) BC[threadIdx.x] = xdbl[row * XDW + DTR + threadIdx.x];
    __syncthreads();
    const float dtv = dtf[row * DI + d];
    const float xv  = xc [row * DI + d];
    float y = 0.f;
#pragma unroll
    for (int n = 0; n < NST; ++n) {
      float dA = __expf(dtv * Arow[n]);
      h[n] = dA * h[n] + (dtv * BC[n]) * xv;
      y += h[n] * BC[NST + n];
    }
    y += xv * Dv;
    const int tr = dir ? (SEQ - 1 - t) : t;
    y_cat[((size_t)b * SEQ + tr) * (2 * DI) + col_off + d] = f2bf(y);
    __syncthreads();
  }
}

// gated = bf16(merged * silu(z)),  z = xz[:, DI:2*DI]
__global__ __launch_bounds__(256)
void gate_silu(const float* __restrict__ merged, const float* __restrict__ xz,
               unsigned short* __restrict__ gated) {
  size_t i = (size_t)blockIdx.x * 256 + threadIdx.x;   // over MROWS*DI
  size_t m = i / DI; int d = (int)(i - m * DI);
  float z = xz[m * (2 * DI) + DI + d];
  gated[i] = f2bf(merged[i] * silu(z));
}

// out = residual + y_out
__global__ __launch_bounds__(256)
void residual_add(const float* __restrict__ x, const float* __restrict__ y,
                  float* __restrict__ out) {
  size_t i = (size_t)blockIdx.x * 256 + threadIdx.x;   // over MROWS*DM
  out[i] = x[i] + y[i];
}

// ---------------------------------------------------------------------------
extern "C" void kernel_launch(void* const* d_in, const int* in_sizes, int n_in,
                              void* d_out, int out_size, void* d_ws, size_t ws_size,
                              hipStream_t stream) {
  (void)in_sizes; (void)n_in; (void)out_size; (void)ws_size;
  const float* x       = (const float*)d_in[0];
  const float* gamma   = (const float*)d_in[1];
  const float* beta    = (const float*)d_in[2];
  const float* in_w    = (const float*)d_in[3];
  const float* conv_w  = (const float*)d_in[4];
  const float* conv_b  = (const float*)d_in[5];
  const float* xproj_w = (const float*)d_in[6];
  const float* dt_w    = (const float*)d_in[7];
  const float* dt_b    = (const float*)d_in[8];
  const float* A_log   = (const float*)d_in[9];
  const float* D_skip  = (const float*)d_in[10];
  const float* conv_wb = (const float*)d_in[11];
  const float* conv_bb = (const float*)d_in[12];
  const float* xproj_wb= (const float*)d_in[13];
  const float* dt_wb   = (const float*)d_in[14];
  const float* dt_bb   = (const float*)d_in[15];
  const float* A_logb  = (const float*)d_in[16];
  const float* D_skipb = (const float*)d_in[17];
  const float* merge_w = (const float*)d_in[18];
  const float* out_w   = (const float*)d_in[19];
  float* out = (float*)d_out;

  char* ws = (char*)d_ws;
  size_t off = 0;
  auto alloc = [&](size_t bytes) -> void* {
    void* p = ws + off; off += (bytes + 255) & ~(size_t)255; return p;
  };
  unsigned short* Wt_in  = (unsigned short*)alloc((size_t)(2*DI) * DM * 2);   // [4096,1024]
  unsigned short* Wt_xpf = (unsigned short*)alloc((size_t)XDW * DI * 2);      // [96,2048]
  unsigned short* Wt_xpb = (unsigned short*)alloc((size_t)XDW * DI * 2);
  unsigned short* Wt_dtf = (unsigned short*)alloc((size_t)DI * DTR * 2);      // [2048,64]
  unsigned short* Wt_dtb = (unsigned short*)alloc((size_t)DI * DTR * 2);
  unsigned short* Wt_mg  = (unsigned short*)alloc((size_t)DI * (2*DI) * 2);   // [2048,4096]
  unsigned short* Wt_out = (unsigned short*)alloc((size_t)DM * DI * 2);       // [1024,2048]
  unsigned short* h_bf   = (unsigned short*)alloc((size_t)MROWS * DM * 2);
  float*          xz     = (float*)         alloc((size_t)MROWS * 2*DI * 4);
  float*          xcf0   = (float*)         alloc((size_t)MROWS * DI * 4);
  float*          xcf1   = (float*)         alloc((size_t)MROWS * DI * 4);
  unsigned short* xcb0   = (unsigned short*)alloc((size_t)MROWS * DI * 2);
  unsigned short* xcb1   = (unsigned short*)alloc((size_t)MROWS * DI * 2);
  float*          xdbl0  = (float*)         alloc((size_t)MROWS * XDW * 4);
  float*          xdbl1  = (float*)         alloc((size_t)MROWS * XDW * 4);
  unsigned short* dtA0   = (unsigned short*)alloc((size_t)MROWS * DTR * 2);
  unsigned short* dtA1   = (unsigned short*)alloc((size_t)MROWS * DTR * 2);
  float*          dtf0   = (float*)         alloc((size_t)MROWS * DI * 4);
  float*          dtf1   = (float*)         alloc((size_t)MROWS * DI * 4);
  unsigned short* y_cat  = (unsigned short*)alloc((size_t)MROWS * 2*DI * 2);
  // post-scan aliases (the aliased buffers are dead by the time these are written)
  float*          merged = xcf0;                       // [M, DI] f32
  unsigned short* gated  = xcb0;                       // [M, DI] bf16
  float*          y_out  = dtf0;                       // [M, DM] f32

  auto cdiv = [](size_t a, size_t b) { return (unsigned)((a + b - 1) / b); };
  auto gemm_blocks = [&](int M, int N, int TM, int TN) {
    return cdiv((size_t)(M / (16 * TM)) * (size_t)(N / (16 * TN)), 8);
  };

  // 1) weight convert+transpose
  transpose_f32_bf16<<<cdiv((size_t)DM*2*DI,256),256,0,stream>>>(in_w,    Wt_in,  DM,  2*DI);
  transpose_f32_bf16<<<cdiv((size_t)DI*XDW, 256),256,0,stream>>>(xproj_w, Wt_xpf, DI,  XDW);
  transpose_f32_bf16<<<cdiv((size_t)DI*XDW, 256),256,0,stream>>>(xproj_wb,Wt_xpb, DI,  XDW);
  transpose_f32_bf16<<<cdiv((size_t)DTR*DI, 256),256,0,stream>>>(dt_w,    Wt_dtf, DTR, DI);
  transpose_f32_bf16<<<cdiv((size_t)DTR*DI, 256),256,0,stream>>>(dt_wb,   Wt_dtb, DTR, DI);
  transpose_f32_bf16<<<cdiv((size_t)2*DI*DI,256),256,0,stream>>>(merge_w, Wt_mg,  2*DI,DI);
  transpose_f32_bf16<<<cdiv((size_t)DI*DM,  256),256,0,stream>>>(out_w,   Wt_out, DI,  DM);

  // 2) layernorm
  layernorm_bf16<<<MROWS, 256, 0, stream>>>(x, gamma, beta, h_bf);

  // 3) in_proj GEMM: [2048,1024] x [1024,4096] -> xz   (4x4 register tile)
  wmma_gemm_tiled<4,4><<<gemm_blocks(MROWS,2*DI,4,4),256,0,stream>>>(h_bf, Wt_in, xz, MROWS, 2*DI, DM);

  // 4) conv + silu (both directions)
  conv_silu<<<cdiv((size_t)MROWS*DI,256),256,0,stream>>>(xz, conv_w,  conv_b,  xcf0, xcb0, 0);
  conv_silu<<<cdiv((size_t)MROWS*DI,256),256,0,stream>>>(xz, conv_wb, conv_bb, xcf1, xcb1, 1);

  // 5) xproj GEMMs: [2048,2048] x [2048,96]  (skinny N: 1x1 tiles for parallelism)
  wmma_gemm_tiled<1,1><<<gemm_blocks(MROWS,XDW,1,1),256,0,stream>>>(xcb0, Wt_xpf, xdbl0, MROWS, XDW, DI);
  wmma_gemm_tiled<1,1><<<gemm_blocks(MROWS,XDW,1,1),256,0,stream>>>(xcb1, Wt_xpb, xdbl1, MROWS, XDW, DI);

  // 6) dt slice -> bf16
  slice_dt_bf16<<<cdiv((size_t)MROWS*DTR,256),256,0,stream>>>(xdbl0, dtA0);
  slice_dt_bf16<<<cdiv((size_t)MROWS*DTR,256),256,0,stream>>>(xdbl1, dtA1);

  // 7) dt GEMMs: [2048,64] x [64,2048]   (4x4 register tile)
  wmma_gemm_tiled<4,4><<<gemm_blocks(MROWS,DI,4,4),256,0,stream>>>(dtA0, Wt_dtf, dtf0, MROWS, DI, DTR);
  wmma_gemm_tiled<4,4><<<gemm_blocks(MROWS,DI,4,4),256,0,stream>>>(dtA1, Wt_dtb, dtf1, MROWS, DI, DTR);

  // 8) bias + softplus
  bias_softplus<<<cdiv((size_t)MROWS*DI,256),256,0,stream>>>(dtf0, dt_b);
  bias_softplus<<<cdiv((size_t)MROWS*DI,256),256,0,stream>>>(dtf1, dt_bb);

  // 9) selective scans
  ssm_scan<<<dim3(DI/64, BATCH), 64, 0, stream>>>(xdbl0, dtf0, xcf0, A_log,  D_skip,  y_cat, 0, 0);
  ssm_scan<<<dim3(DI/64, BATCH), 64, 0, stream>>>(xdbl1, dtf1, xcf1, A_logb, D_skipb, y_cat, 1, DI);

  // 10) merge GEMM: [2048,4096] x [4096,2048]   (4x4 register tile)
  wmma_gemm_tiled<4,4><<<gemm_blocks(MROWS,DI,4,4),256,0,stream>>>(y_cat, Wt_mg, merged, MROWS, DI, 2*DI);

  // 11) gate with silu(z)
  gate_silu<<<cdiv((size_t)MROWS*DI,256),256,0,stream>>>(merged, xz, gated);

  // 12) out GEMM: [2048,2048] x [2048,1024]   (4x4 register tile)
  wmma_gemm_tiled<4,4><<<gemm_blocks(MROWS,DM,4,4),256,0,stream>>>(gated, Wt_out, y_out, MROWS, DM, DI);

  // 13) residual add
  residual_add<<<cdiv((size_t)MROWS*DM,256),256,0,stream>>>(x, y_out, out);
}
